// Block_27187142983954
// MI455X (gfx1250) — matrix operand order repaired and checked by reading, not compiled, not run on hardware
//
#include <hip/hip_runtime.h>
#include <hip/hip_bf16.h>
#include <stdint.h>

typedef __attribute__((ext_vector_type(16))) __bf16   v16bf;
typedef __attribute__((ext_vector_type(8)))  float    v8f;
typedef __attribute__((ext_vector_type(8)))  uint16_t us8;
typedef __attribute__((ext_vector_type(16))) uint16_t us16;

// ---------------- bf16 helpers (RNE) ----------------
static __device__ __forceinline__ uint16_t f2bf(float f) {
    uint32_t u = __builtin_bit_cast(uint32_t, f);
    u += 0x7FFFu + ((u >> 16) & 1u);
    return (uint16_t)(u >> 16);
}
static __device__ __forceinline__ float bf2f(uint16_t h) {
    return __builtin_bit_cast(float, (uint32_t)h << 16);
}
// Build a WMMA 16-element bf16 fragment from two contiguous 8-element LDS runs.
static __device__ __forceinline__ v16bf frag16(const uint16_t* p0, const uint16_t* p1) {
    us8 lo = *(const us8*)p0;
    us8 hi = *(const us8*)p1;
    us16 c = __builtin_shufflevector(lo, hi, 0,1,2,3,4,5,6,7,8,9,10,11,12,13,14,15);
    return __builtin_bit_cast(v16bf, c);
}

// ---------------- CDNA5 async global->LDS DMA (ASYNCcnt-tracked) ----------------
// LDS operand = byte offset within the workgroup LDS allocation = low 32 bits of the
// generic shared pointer (flat->LDS address truncation, ISA 10.2).
static __device__ __forceinline__ void async_ld16(uint16_t* lds_dst, const uint16_t* gsrc) {
    uint32_t loff = (uint32_t)(uintptr_t)lds_dst;
    uint64_t ga   = (uint64_t)(uintptr_t)gsrc;
    asm volatile("global_load_async_to_lds_b128 %0, %1, off"
                 :: "v"(loff), "v"(ga) : "memory");
}
static __device__ __forceinline__ void wait_async0() {
    asm volatile("s_wait_asynccnt 0" ::: "memory");
}

// ---------------- fp32 -> bf16 convert ----------------
__global__ void __launch_bounds__(256) cvt_kernel(const float* __restrict__ src,
                                                  uint16_t* __restrict__ dst, long n) {
    long i = (long)blockIdx.x * blockDim.x + threadIdx.x;
    long stride = (long)gridDim.x * blockDim.x;
    for (; i < n; i += stride) dst[i] = f2bf(src[i]);
}

// ---------------- QKV GEMM: [N,1024] @ [3072,1024]^T + bias -> bf16 [N,3072] ----------------
// Tile 128x128, BK=32, double-buffered async LDS staging.
// 8 waves: 2(M) x 4(N), each wave 64x32 -> 4x2 WMMA tiles.
static __device__ __forceinline__ void qkv_issue(const uint16_t* xbf, const uint16_t* w1,
                                                 uint16_t* sA, uint16_t* sB,
                                                 int m0, int n0, int k0, int tid) {
    int row = tid >> 1, seg = tid & 1;
    const uint16_t* ga = xbf + (size_t)(m0 + row) * 1024 + k0 + seg * 16;
    uint16_t* la = sA + row * 32 + seg * 16;
    async_ld16(la, ga);
    async_ld16(la + 8, ga + 8);
    const uint16_t* gb = w1 + (size_t)(n0 + row) * 1024 + k0 + seg * 16;
    uint16_t* lb = sB + row * 32 + seg * 16;
    async_ld16(lb, gb);
    async_ld16(lb + 8, gb + 8);
}

__global__ void __launch_bounds__(256) gemm_qkv_kernel(const uint16_t* __restrict__ xbf,
                                                       const uint16_t* __restrict__ w1,
                                                       const float* __restrict__ qkb,
                                                       const float* __restrict__ vb,
                                                       uint16_t* __restrict__ qkv) {
    extern __shared__ __align__(16) char smem[];
    uint16_t* sA = (uint16_t*)smem;                    // 2 x (128 x 32)
    uint16_t* sB = (uint16_t*)(smem + 2 * 128*32*2);   // 2 x (128 x 32)

    const int tid  = threadIdx.x;
    const int wv   = tid >> 5;
    const int lane = tid & 31;
    const int ph   = lane >> 4;       // half-wave select
    const int l16  = lane & 15;
    const int m0   = blockIdx.y * 128;
    const int n0   = blockIdx.x * 128;
    const int wm   = (wv >> 2) * 64;  // wave row offset
    const int wn   = (wv & 3) * 32;   // wave col offset

    v8f acc[4][2];
    #pragma unroll
    for (int i = 0; i < 4; ++i)
        #pragma unroll
        for (int j = 0; j < 2; ++j) acc[i][j] = 0.0f;

    qkv_issue(xbf, w1, sA, sB, m0, n0, 0, tid);   // prime buffer 0

    for (int it = 0; it < 32; ++it) {
        const int p = it & 1;
        uint16_t* cA = sA + p * (128*32);
        uint16_t* cB = sB + p * (128*32);
        wait_async0();           // my async copies into buffer p done
        __syncthreads();         // everyone's done; prev-buffer reads retired
        if (it + 1 < 32)
            qkv_issue(xbf, w1, sA + (p ^ 1) * (128*32), sB + (p ^ 1) * (128*32),
                      m0, n0, (it + 1) * 32, tid);
        v16bf aF[4];
        #pragma unroll
        for (int ms = 0; ms < 4; ++ms) {
            int r = wm + ms*16 + l16;
            aF[ms] = frag16(cA + r*32 + ph*8, cA + r*32 + 16 + ph*8);
        }
        #pragma unroll
        for (int ns = 0; ns < 2; ++ns) {
            int c = wn + ns*16 + l16;
            v16bf bF = frag16(cB + c*32 + ph*8, cB + c*32 + 16 + ph*8);
            #pragma unroll
            for (int ms = 0; ms < 4; ++ms)
                acc[ms][ns] = __builtin_amdgcn_wmma_f32_16x16x32_bf16(
                    false, aF[ms], false, bF, (short)0, acc[ms][ns], false, false);
        }
    }

    #pragma unroll
    for (int ms = 0; ms < 4; ++ms)
        #pragma unroll
        for (int ns = 0; ns < 2; ++ns)
            #pragma unroll
            for (int r = 0; r < 8; ++r) {
                int grow = m0 + wm + ms*16 + ph*8 + r;
                int gcol = n0 + wn + ns*16 + l16;
                float b = (gcol < 2048) ? qkb[gcol] : vb[gcol - 2048];
                qkv[(size_t)grow*3072 + gcol] = f2bf(acc[ms][ns][r] + b);
            }
}

// ---------------- per-token head attention: 16x16 softmax + (16x16)@(16x64) ----------------
// One wave per token, 8 tokens per block staged through LDS via async DMA.
__global__ void __launch_bounds__(256) attn_kernel(const uint16_t* __restrict__ qkv,
                                                   uint16_t* __restrict__ outp) {
    extern __shared__ __align__(16) char smem[];
    uint16_t* sT = (uint16_t*)smem;              // 8 tokens * 3072 bf16
    float*    sP = (float*)(smem + 8*3072*2);    // 8 * 16*16 probs

    const int tid  = threadIdx.x;
    const int wv   = tid >> 5;
    const int lane = tid & 31;
    const int ph   = lane >> 4;
    const int l16  = lane & 15;
    const size_t tok0 = (size_t)blockIdx.x * 8;

    const uint16_t* gsrc = qkv + tok0 * 3072;
    #pragma unroll
    for (int i = 0; i < 12; ++i) {
        int idx = (tid + i * 256) * 8;           // element offset, 16B chunks
        async_ld16(sT + idx, gsrc + idx);
    }
    wait_async0();
    __syncthreads();

    const uint16_t* q = sT + wv*3072;
    const uint16_t* k = q + 1024;
    const uint16_t* v = q + 2048;
    float* probs = sP + wv*256;

    // scores + softmax: lane owns rows h=ph*8+r, column g=l16; row lives in a 16-lane group
    #pragma unroll
    for (int r = 0; r < 8; ++r) {
        int h = ph*8 + r;
        float s = 0.f;
        for (int d = 0; d < 64; ++d)
            s += bf2f(q[h*64 + d]) * bf2f(k[l16*64 + d]);
        s *= 0.125f;                 // 1/sqrt(64)
        float m = s;
        for (int msk = 1; msk < 16; msk <<= 1) m = fmaxf(m, __shfl_xor(m, msk, 32));
        float e = __expf(s - m);
        float sum = e;
        for (int msk = 1; msk < 16; msk <<= 1) sum += __shfl_xor(sum, msk, 32);
        probs[h*16 + l16] = e / sum;
    }
    __syncthreads();

    // out[h][d] = sum_g p[h][g] * v[g][d]; lane owns h=lane/2, d in [ (lane&1)*32, +32 )
    const int h2 = lane >> 1;
    const int d0 = (lane & 1) * 32;
    float o[32];
    #pragma unroll
    for (int j = 0; j < 32; ++j) o[j] = 0.f;
    for (int g = 0; g < 16; ++g) {
        float p = probs[h2*16 + g];
        #pragma unroll
        for (int j = 0; j < 32; ++j) o[j] += p * bf2f(v[g*64 + d0 + j]);
    }
    uint16_t* od = outp + (tok0 + wv)*1024 + h2*64 + d0;
    #pragma unroll
    for (int j = 0; j < 32; ++j) od[j] = f2bf(o[j]);
}

// ---------------- fused GEMM (+bias [+gelu]) + residual + LayerNorm ----------------
// Tile: M=32 tokens, N=full 1024 (so LN stats are block-local). BK=32,
// double-buffered async LDS staging (B strip = 64KB per K-step).
// 8 waves: wave w owns cols [w*128, w*128+128): 2(M) x 8(N) WMMA tiles each.
static constexpr unsigned LN_ABUF = 32*32;     // elems per A buffer
static constexpr unsigned LN_BBUF = 1024*32;   // elems per B buffer
static constexpr unsigned LN_LDS  = 2*LN_ABUF*2 + 2*LN_BBUF*2 + 8*32*8 + 32*8; // 137472 B

static __device__ __forceinline__ void ln_issue(const uint16_t* act, const uint16_t* w,
                                                uint16_t* sA, uint16_t* sB,
                                                int m0, int k0, int tid) {
    #pragma unroll
    for (int i = 0; i < 16; ++i) {       // B strip: 1024x32
        int idx = tid + i*256;           // us8 chunk index in [0,4096)
        int row = idx >> 2, seg = idx & 3;
        async_ld16(sB + row*32 + seg*8, w + (size_t)row*1024 + k0 + seg*8);
    }
    if (tid < 128) {                     // A tile: 32x32
        int row = tid >> 2, seg = tid & 3;
        async_ld16(sA + row*32 + seg*8, act + (size_t)(m0 + row)*1024 + k0 + seg*8);
    }
}

template <bool GELU, bool OUTF32>
__global__ void __launch_bounds__(256) gemm_ln_kernel(const uint16_t* __restrict__ act,
                                                      const uint16_t* __restrict__ w,
                                                      const float* __restrict__ bias,
                                                      const float* __restrict__ resid,
                                                      const float* __restrict__ lng,
                                                      const float* __restrict__ lnb,
                                                      uint16_t* __restrict__ out_bf,
                                                      float* __restrict__ out_f) {
    extern __shared__ __align__(16) char smem[];
    uint16_t* sA = (uint16_t*)smem;                               // 2 x (32 x 32)
    uint16_t* sB = (uint16_t*)(smem + 2*LN_ABUF*2);               // 2 x (1024 x 32)
    float2*   sWv = (float2*)(smem + 2*LN_ABUF*2 + 2*LN_BBUF*2);  // [8 waves][32 rows]
    float2*   sSt = (float2*)((char*)sWv + 8*32*8);               // [32 rows] (mean, rstd)

    const int tid  = threadIdx.x;
    const int wv   = tid >> 5;
    const int lane = tid & 31;
    const int ph   = lane >> 4;
    const int l16  = lane & 15;
    const int m0   = blockIdx.x * 32;

    v8f acc[2][8];
    #pragma unroll
    for (int i = 0; i < 2; ++i)
        #pragma unroll
        for (int j = 0; j < 8; ++j) acc[i][j] = 0.0f;

    ln_issue(act, w, sA, sB, m0, 0, tid);   // prime buffer 0

    for (int it = 0; it < 32; ++it) {
        const int p = it & 1;
        uint16_t* cA = sA + p * LN_ABUF;
        uint16_t* cB = sB + p * LN_BBUF;
        wait_async0();
        __syncthreads();
        if (it + 1 < 32)
            ln_issue(act, w, sA + (p ^ 1) * LN_ABUF, sB + (p ^ 1) * LN_BBUF,
                     m0, (it + 1) * 32, tid);
        v16bf aF[2];
        #pragma unroll
        for (int ms = 0; ms < 2; ++ms) {
            int r = ms*16 + l16;
            aF[ms] = frag16(cA + r*32 + ph*8, cA + r*32 + 16 + ph*8);
        }
        #pragma unroll
        for (int ns = 0; ns < 8; ++ns) {
            int c = wv*128 + ns*16 + l16;
            v16bf bF = frag16(cB + c*32 + ph*8, cB + c*32 + 16 + ph*8);
            #pragma unroll
            for (int ms = 0; ms < 2; ++ms)
                acc[ms][ns] = __builtin_amdgcn_wmma_f32_16x16x32_bf16(
                    false, aF[ms], false, bF, (short)0, acc[ms][ns], false, false);
        }
    }

    // epilogue: bias [+gelu] + residual, per-row partial stats via 16-lane shuffles
    #pragma unroll
    for (int ms = 0; ms < 2; ++ms)
        #pragma unroll
        for (int r = 0; r < 8; ++r) {
            int rowl = ms*16 + ph*8 + r;
            size_t grow = (size_t)(m0 + rowl);
            float s = 0.f, qq = 0.f;
            #pragma unroll
            for (int ns = 0; ns < 8; ++ns) {
                int col = wv*128 + ns*16 + l16;
                float val = acc[ms][ns][r] + bias[col];
                if (GELU) val = 0.5f * val * (1.0f + erff(val * 0.70710678118654752f));
                val += resid[grow*1024 + col];
                acc[ms][ns][r] = val;
                s += val; qq += val*val;
            }
            for (int msk = 1; msk < 16; msk <<= 1) {
                s  += __shfl_xor(s,  msk, 32);
                qq += __shfl_xor(qq, msk, 32);
            }
            if (l16 == 0) sWv[wv*32 + rowl] = make_float2(s, qq);
        }
    __syncthreads();
    if (tid < 32) {
        float s = 0.f, qq = 0.f;
        #pragma unroll
        for (int w8 = 0; w8 < 8; ++w8) { float2 t = sWv[w8*32 + tid]; s += t.x; qq += t.y; }
        float mu  = s * (1.0f/1024.0f);
        float var = qq * (1.0f/1024.0f) - mu*mu;
        sSt[tid] = make_float2(mu, rsqrtf(var + 1e-5f));
    }
    __syncthreads();
    #pragma unroll
    for (int ms = 0; ms < 2; ++ms)
        #pragma unroll
        for (int r = 0; r < 8; ++r) {
            int rowl = ms*16 + ph*8 + r;
            float2 st = sSt[rowl];
            size_t grow = (size_t)(m0 + rowl);
            #pragma unroll
            for (int ns = 0; ns < 8; ++ns) {
                int col = wv*128 + ns*16 + l16;
                float val = (acc[ms][ns][r] - st.x) * st.y * lng[col] + lnb[col];
                if (OUTF32) out_f[grow*1024 + col] = val;
                else        out_bf[grow*1024 + col] = f2bf(val);
            }
        }
}

// ---------------- host launch ----------------
extern "C" void kernel_launch(void* const* d_in, const int* in_sizes, int n_in,
                              void* d_out, int out_size, void* d_ws, size_t ws_size,
                              hipStream_t stream) {
    (void)in_sizes; (void)n_in; (void)out_size; (void)ws_size;
    const float* x      = (const float*)d_in[0];
    const float* qk_w   = (const float*)d_in[1];
    const float* qk_b   = (const float*)d_in[2];
    const float* v_w    = (const float*)d_in[3];
    const float* v_b    = (const float*)d_in[4];
    const float* proj_w = (const float*)d_in[5];
    const float* proj_b = (const float*)d_in[6];
    const float* ff_w   = (const float*)d_in[7];
    const float* ff_b   = (const float*)d_in[8];
    const float* ln_g   = (const float*)d_in[9];
    const float* ln_b   = (const float*)d_in[10];
    float* out = (float*)d_out;

    const size_t NT = 32768;  // B*S tokens
    uint16_t* xbf  = (uint16_t*)d_ws;
    uint16_t* w1   = xbf  + NT * 1024;            // qk_w (2048) ++ v_w (1024) rows
    uint16_t* wp   = w1   + (size_t)3072 * 1024;
    uint16_t* wf   = wp   + (size_t)1024 * 1024;
    uint16_t* qkvb = wf   + (size_t)1024 * 1024;  // [NT, 3072]
    uint16_t* attb = qkvb + NT * 3072;            // [NT, 1024]
    uint16_t* ybf  = attb + NT * 1024;            // [NT, 1024]

    cvt_kernel<<<2048, 256, 0, stream>>>(x, xbf, (long)(NT * 1024));
    cvt_kernel<<<1024, 256, 0, stream>>>(qk_w, w1, 2048L * 1024);
    cvt_kernel<<<512, 256, 0, stream>>>(v_w, w1 + (size_t)2048 * 1024, 1024L * 1024);
    cvt_kernel<<<512, 256, 0, stream>>>(proj_w, wp, 1024L * 1024);
    cvt_kernel<<<512, 256, 0, stream>>>(ff_w, wf, 1024L * 1024);

    gemm_qkv_kernel<<<dim3(24, 256), 256, 2*(128*32*2)*2, stream>>>(xbf, w1, qk_b, v_b, qkvb);
    attn_kernel<<<4096, 256, 8*3072*2 + 8*256*4, stream>>>(qkvb, attb);
    gemm_ln_kernel<false, false><<<1024, 256, LN_LDS, stream>>>(
        attb, wp, proj_b, x, ln_g, ln_b, ybf, nullptr);
    gemm_ln_kernel<true, true><<<1024, 256, LN_LDS, stream>>>(
        ybf, wf, ff_b, x, ln_g, ln_b, nullptr, out);
}
